// NaiveSubMConv3d_9603546874116
// MI455X (gfx1250) — matrix operand order repaired and checked by reading, not compiled, not run on hardware
//
#include <hip/hip_runtime.h>

typedef __attribute__((ext_vector_type(16))) _Float16 v16h;
typedef __attribute__((ext_vector_type(8)))  _Float16 v8h;
typedef __attribute__((ext_vector_type(8)))  float    v8f;

#define B_   2
#define H0_  64
#define H1_  256
#define H2_  256
#define N_   100000
#define C_   64
#define K_   64
#define TBL_ (B_*H0_*H1_*H2_)               // 8,388,608 entries (32 MB)
#define TAP_HALVES   (4*2*32*16)             // 4096 f16 = 8 KB per tap
#define WPACK_HALVES (27*TAP_HALVES)         // 110,592 f16 = 221,184 B
#define TABLE_OFS  (1u<<18)                  // ws + 256 KB
#define FEAT16_OFS (TABLE_OFS + (size_t)TBL_*4)   // ws + 33.8 MB
#define F16_ROWS   (N_ + 16)                 // +1 zero row (padded to 16)

// ---------------- prep kernels ----------------

__global__ void init_table_k(int* __restrict__ table) {
    int i = blockIdx.x * blockDim.x + threadIdx.x;
    if (i < TBL_) table[i] = -1;
}

__global__ void build_table_k(const int* __restrict__ idx, int* __restrict__ table) {
    int n = blockIdx.x * blockDim.x + threadIdx.x;
    if (n < N_) {
        int b = idx[n*4+0], z = idx[n*4+1], y = idx[n*4+2], x = idx[n*4+3];
        int flat = ((b*H0_ + z)*H1_ + y)*H2_ + x;
        table[flat] = n;
    }
}

// features f32 -> f16 once; rows [N_, F16_ROWS) zeroed (inactive-neighbor target)
__global__ void cvt_feats_k(const float* __restrict__ f, _Float16* __restrict__ f16) {
    int t = blockIdx.x * blockDim.x + threadIdx.x;
    if (t >= F16_ROWS * C_) return;
    f16[t] = (t < N_ * C_) ? (_Float16)f[t] : (_Float16)0.0f;
}

// weight (K,3,3,3,C) f32 -> wpack[off][nt][ks][lane][i] f16 in WMMA B 32x16 layout
__global__ void pack_weights_k(const float* __restrict__ w, _Float16* __restrict__ wpack) {
    int t = blockIdx.x * blockDim.x + threadIdx.x;
    if (t >= WPACK_HALVES) return;
    int i    =  t        & 15;
    int lane = (t >> 4)  & 31;
    int ks   = (t >> 9)  & 1;
    int nt   = (t >> 10) & 3;
    int off  =  t >> 12;                      // 0..26 (rz*9 + ry*3 + rx)
    int c    = ks*32 + (i >> 3)*16 + (lane >> 4)*8 + (i & 7);
    int kout = nt*16 + (lane & 15);
    wpack[t] = (_Float16)w[(kout*27 + off)*C_ + c];
}

// ---------------- helpers ----------------

static __device__ __forceinline__ void wait_async0() {
#if __has_builtin(__builtin_amdgcn_s_wait_asynccnt)
    __builtin_amdgcn_s_wait_asynccnt(0);
#else
    asm volatile("s_wait_asynccnt 0x0" ::: "memory");
#endif
}

// ---------------- main conv kernel ----------------
// 5 waves / block (160 thr), one wave32 per 32-point tile -> 625 blocks x 5 = 3125 exact.
// Per tap: B block (8 KB) staged in LDS via async global->LDS copy, double-buffered;
// 16 WMMAs per tap (4 N-tiles x 2 K-slices x 2 M sub-tiles), B read via ds_load.

__global__ __launch_bounds__(160) void subm_conv_wmma_k(
    const _Float16* __restrict__ feats16,
    const int*      __restrict__ idx,
    const _Float16* __restrict__ wpack,
    const float*    __restrict__ bias,
    const int*      __restrict__ table,
    float*          __restrict__ out)
{
    __shared__ __align__(64) _Float16 bsh[2][TAP_HALVES];   // 2 x 8 KB

    const int tid  = threadIdx.x;
    const int lane = tid & 31;
    const int wave = blockIdx.x * 5 + (tid >> 5);            // 0..3124, all active
    const int row  = lane & 15;
    const int g    = lane >> 4;
    const int m0   = wave << 5;

    // async fill of one tap's B block: 512 x 16B chunks spread over 160 threads
    auto prefetch_tap = [&](int off, int buf) {
        const _Float16* gsrc = wpack + (size_t)off * TAP_HALVES;
        for (int c = tid; c < TAP_HALVES/8; c += 160) {
            unsigned int  lds = (unsigned int)(unsigned long long)(&bsh[buf][c * 8]);
            unsigned long long ga = (unsigned long long)(gsrc + c * 8);
            asm volatile("global_load_async_to_lds_b128 %0, %1, off"
                         :: "v"(lds), "v"(ga) : "memory");
        }
    };

    const int4 co0 = ((const int4*)idx)[m0 + row];
    const int4 co1 = ((const int4*)idx)[m0 + 16 + row];

    v8f acc[2][4];
    #pragma unroll
    for (int t = 0; t < 2; ++t)
        #pragma unroll
        for (int nt = 0; nt < 4; ++nt)
            acc[t][nt] = (v8f){};

    prefetch_tap(0, 0);
    wait_async0();
    __syncthreads();

    #pragma unroll 1
    for (int off = 0; off < 27; ++off) {
        const int cur = off & 1;
        if (off + 1 < 27) prefetch_tap(off + 1, cur ^ 1);   // overlaps compute

        const int dz = off / 9 - 1;
        const int dy = (off / 3) % 3 - 1;
        const int dx = off % 3 - 1;

        const _Float16* src[2];
        #pragma unroll
        for (int t = 0; t < 2; ++t) {
            const int4 co = t ? co1 : co0;
            const int nz = co.y + dz, ny = co.z + dy, nx = co.w + dx;
            const bool inb = ((unsigned)nz < (unsigned)H0_) &
                             ((unsigned)ny < (unsigned)H1_) &
                             ((unsigned)nx < (unsigned)H2_);
            int nbr = -1;
            if (inb) {
                const int flat = ((co.x*H0_ + nz)*H1_ + ny)*H2_ + nx;
                nbr = table[flat];
            }
            src[t] = feats16 + (size_t)(nbr >= 0 ? nbr : N_) * C_;  // miss -> zero row
        }

        // A fragments: ISA 16-bit A 16x32 layout, straight f16 loads
        v16h a[2][2];
        #pragma unroll
        for (int t = 0; t < 2; ++t)
            #pragma unroll
            for (int ks = 0; ks < 2; ++ks) {
                const _Float16* p = src[t] + ks*32 + g*8;
                v8h lo = *(const v8h*)(p);
                v8h hi = *(const v8h*)(p + 16);
                a[t][ks] = __builtin_shufflevector(lo, hi,
                    0,1,2,3,4,5,6,7,8,9,10,11,12,13,14,15);
            }

        // 16 WMMAs; B fragments from LDS (ds_load), reused across both M sub-tiles
        #pragma unroll
        for (int nt = 0; nt < 4; ++nt) {
            #pragma unroll
            for (int ks = 0; ks < 2; ++ks) {
                v16h bfrag = *(const v16h*)&bsh[cur][((nt*2 + ks)*32 + lane)*16];
                #pragma unroll
                for (int t = 0; t < 2; ++t) {
                    acc[t][nt] = __builtin_amdgcn_wmma_f32_16x16x32_f16(
                        false, a[t][ks], false, bfrag,
                        (short)0, acc[t][nt], false, false);
                }
            }
        }

        wait_async0();        // our prefetch for off+1 landed in LDS
        __syncthreads();      // publish across the block's 5 waves
    }

    // epilogue: C/D 16x16 f32 layout -> out[n][kout] row-major, + bias
    #pragma unroll
    for (int t = 0; t < 2; ++t) {
        #pragma unroll
        for (int nt = 0; nt < 4; ++nt) {
            const int kout = nt*16 + row;
            const float bv = bias[kout];
            #pragma unroll
            for (int v = 0; v < 8; ++v) {
                const int n = m0 + t*16 + g*8 + v;
                out[(size_t)n * K_ + kout] = acc[t][nt][v] + bv;
            }
        }
    }
}

// ---------------- launch ----------------

extern "C" void kernel_launch(void* const* d_in, const int* in_sizes, int n_in,
                              void* d_out, int out_size, void* d_ws, size_t ws_size,
                              hipStream_t stream) {
    (void)in_sizes; (void)n_in; (void)out_size; (void)ws_size;
    const float* feats = (const float*)d_in[0];   // (N, 64) f32
    const int*   idx   = (const int*)  d_in[1];   // (N, 4)  i32
    const float* w     = (const float*)d_in[2];   // (64,3,3,3,64) f32
    const float* bias  = (const float*)d_in[3];   // (64,) f32
    float*       out   = (float*)d_out;           // (N, 64) f32

    _Float16* wpack   = (_Float16*)d_ws;                        // 221 KB
    int*      table   = (int*)((char*)d_ws + TABLE_OFS);        // 32 MB
    _Float16* feats16 = (_Float16*)((char*)d_ws + FEAT16_OFS);  // 12.8 MB

    init_table_k  <<<(TBL_ + 255)/256, 256, 0, stream>>>(table);
    build_table_k <<<(N_   + 255)/256, 256, 0, stream>>>(idx, table);
    cvt_feats_k   <<<(F16_ROWS*C_ + 255)/256, 256, 0, stream>>>(feats, feats16);
    pack_weights_k<<<(WPACK_HALVES + 255)/256, 256, 0, stream>>>(w, wpack);

    const int blocks = (N_ >> 5) / 5;             // 3125 tiles / 5 waves = 625
    subm_conv_wmma_k<<<blocks, 160, 0, stream>>>(feats16, idx, wpack, bias, table, out);
}
